// get_fingerprint_modulated_40587440947385
// MI455X (gfx1250) — matrix operand order repaired and checked by reading, not compiled, not run on hardware
//
#include <hip/hip_runtime.h>
#include <hip/hip_bf16.h>
#include <math.h>

typedef _Float16 v16h __attribute__((ext_vector_type(16)));
typedef float    v8f  __attribute__((ext_vector_type(8)));

#define B_    8
#define OUT_  512
#define IN_   512
#define H_    64
#define W_    64
#define LAT_  512
#define FIN_  128
#define FANIN 4608               // IN_*3*3
#define NW_B  (OUT_*FANIN)       // 2359296 elements per batch
#define WSCALE (1.0f/48.0f)      // sqrt(2/4608)
#define HP    66                 // padded H
#define WP    66                 // padded W
#define ASZ   7680               // one A buffer: 3*64*40 halfs
#define BSZ   2640               // one B buffer: 66*40 halfs

__device__ __forceinline__ float relu6f(float x){ return fminf(fmaxf(x,0.f),6.f); }
__device__ __forceinline__ float lreluf(float x){ return x>=0.f ? x : 0.01f*x; }

// issue one 16B global -> LDS async copy (ASYNCcnt-tracked, no VGPR round trip)
__device__ __forceinline__ void async_cp16(const _Float16* gsrc, _Float16* lds_dst){
    unsigned lds_addr = (unsigned)(unsigned long long)lds_dst;  // low 32b = LDS offset
    asm volatile("global_load_async_to_lds_b128 %0, %1, off"
                 :: "v"(lds_addr), "v"(gsrc) : "memory");
}
__device__ __forceinline__ void async_wait0(){
    asm volatile("s_wait_asynccnt 0x0" ::: "memory");
}

// ---------------- small prep kernels ----------------

__global__ void k_init(float* acc){
    if (threadIdx.x < 32) acc[threadIdx.x] = 0.f;
}

__global__ void k_zero(uint4* __restrict__ p){
    p[(size_t)blockIdx.x*256 + threadIdx.x] = uint4{0,0,0,0};
}

// input [B][IN][64][64] f32 -> padded channels-last [B][66][66][512] f16 (interior only)
__global__ __launch_bounds__(256) void k_pad_cvt(const float* __restrict__ in,
                                                 _Float16* __restrict__ Xp){
    __shared__ __align__(16) _Float16 T[64*40];
    const int y = blockIdx.x, b = blockIdx.y;
    const int tid = threadIdx.x;
    const int il = tid >> 3, xs = (tid & 7)*8;   // load phase: 32 ch x 64 x
    const int xw = tid >> 2, sg = tid & 3;       // store phase: 64 x x 4 segs of 8 ch
    for (int ib = 0; ib < 16; ++ib){
        const int i0 = ib*32;
        __syncthreads();
        const float* src = in + ((size_t)(b*IN_ + i0 + il)*H_ + y)*W_ + xs;
#pragma unroll
        for (int j = 0; j < 8; ++j) T[(xs + j)*40 + il] = (_Float16)src[j];
        __syncthreads();
        uint4 v = *(const uint4*)&T[xw*40 + sg*8];
        *(uint4*)&Xp[(((size_t)(b*HP + y + 1))*WP + (xw + 1))*512 + i0 + sg*8] = v;
    }
}

// s_mod[b][i], h1[b][j], t1[b][o]   (grid 36x256 = 9216 threads)
__global__ void k_small1(const float* __restrict__ latent, const float* __restrict__ W_mlp,
                         const float* __restrict__ b_mlp,  const float* __restrict__ fp,
                         const float* __restrict__ W_fin,  const float* __restrict__ Wb1,
                         float* __restrict__ smod, float* __restrict__ h1, float* __restrict__ t1){
    int tid = blockIdx.x*256 + threadIdx.x;
    if (tid < 4096){
        int b = tid >> 9, i = tid & 511;
        float acc = 0.f;
        for (int l=0;l<LAT_;++l) acc += latent[b*LAT_+l]*W_mlp[i*LAT_+l];
        smod[tid] = 1.f + acc + b_mlp[i];
    } else if (tid < 5120){
        int t = tid-4096; int b = t >> 7, j = t & 127;
        float acc = 0.f;
        for (int l=0;l<FIN_;++l) acc += fp[b*FIN_+l]*W_fin[j*FIN_+l];
        h1[t] = lreluf(acc);
    } else { // tid < 9216
        int t = tid-5120; int b = t >> 9, o = t & 511;
        float acc = 0.f;
        for (int l=0;l<FIN_;++l) acc += fp[b*FIN_+l]*Wb1[o*FIN_+l];
        t1[t] = lreluf(acc);
    }
}

// style[b][9216], bias_total[b][o]   (grid 304x256 = 77824 threads)
__global__ void k_small2(const float* __restrict__ h1, const float* __restrict__ W_style,
                         const float* __restrict__ b_style, const float* __restrict__ t1,
                         const float* __restrict__ Wb2, const float* __restrict__ ori_bias,
                         const float* __restrict__ b_mask,
                         float* __restrict__ style, float* __restrict__ biasT){
    int tid = blockIdx.x*256 + threadIdx.x;
    if (tid < 73728){
        int b = tid / 9216, n = tid - b*9216;
        float acc = 0.f;
        for (int l=0;l<128;++l) acc += h1[b*128+l]*W_style[n*128+l];
        style[tid] = acc + b_style[n];
    } else {
        int t = tid-73728; int b = t >> 9, o = t & 511;
        float acc = 0.f;
        for (int l=0;l<OUT_;++l) acc += t1[b*OUT_+l]*Wb2[o*OUT_+l];
        float sb = lreluf(acc);
        float bm = relu6f(b_mask[o]);
        biasT[t] = ori_bias[o]*bm*(1.f/6.f) + (6.f-bm)*(1.f/6.f)*sb;
    }
}

// mean/std of scaled base weight -> acc[0]=sum acc[1]=sumsq
__global__ void k_base_stats(const float* __restrict__ ow, float* __restrict__ acc){
    __shared__ float sA[256], sB[256];
    int tid = threadIdx.x;
    float s=0.f, q=0.f;
    for (int idx = blockIdx.x*256 + tid; idx < NW_B; idx += 1024*256){
        float v = ow[idx]*WSCALE; s += v; q += v*v;
    }
    sA[tid]=s; sB[tid]=q; __syncthreads();
    for (int st=128; st>0; st>>=1){ if (tid<st){ sA[tid]+=sA[tid+st]; sB[tid]+=sB[tid+st]; } __syncthreads(); }
    if (!tid){ atomicAdd(&acc[0], sA[0]); atomicAdd(&acc[1], sB[0]); }
}

// d[b][o] = rsqrt(sum_{i,tap} (smod*base)^2 + 1e-5)   grid 4096 blocks
__global__ void k_demod(const float* __restrict__ ow, const float* __restrict__ smod,
                        float* __restrict__ dvec){
    __shared__ float sA[256];
    int b = blockIdx.x >> 9, o = blockIdx.x & 511, tid = threadIdx.x;
    float s = 0.f;
    for (int t = tid; t < FANIN; t += 256){
        int i = t/9;
        float v = smod[b*IN_+i]*ow[o*FANIN+t]*WSCALE;
        s += v*v;
    }
    sA[tid]=s; __syncthreads();
    for (int st=128; st>0; st>>=1){ if (tid<st) sA[tid]+=sA[tid+st]; __syncthreads(); }
    if (!tid) dvec[blockIdx.x] = rsqrtf(sA[0] + 1e-5f);
}

__device__ __forceinline__ float compute_nw(int b, int idx,
        const float* ow, const float* fwt, const float* wmask,
        const float* smod, const float* style, const float* dvec,
        float* w_out, float* fw_out){
    int o = idx / FANIN; int rem = idx - o*FANIN; int i = rem/9; int tap = rem - i*9;
    float base = ow[idx]*WSCALE;
    float w = dvec[b*OUT_+o]*smod[b*IN_+i]*base;
    float sc = style[b*9216 + i*9 + tap];
    float sh = style[b*9216 + 4608 + i*9 + tap];
    float sw = fwt[idx]*(sc+1.f) + sh;
    float fw = (6.f - relu6f(wmask[o*IN_+i]))*(1.f/6.f);
    *w_out = w; *fw_out = fw;
    return (1.f-fw)*w + fw*sw;
}

// per-batch sum/sumsq of nw -> acc[2+2b], acc[3+2b]   grid (288, 8)
__global__ void k_w_stats(const float* __restrict__ ow, const float* __restrict__ fwt,
                          const float* __restrict__ wmask, const float* __restrict__ smod,
                          const float* __restrict__ style, const float* __restrict__ dvec,
                          float* __restrict__ acc){
    __shared__ float sA[256], sB[256];
    int b = blockIdx.y, tid = threadIdx.x;
    float s=0.f, q=0.f;
    for (int k=0;k<32;++k){
        int idx = (k*288 + blockIdx.x)*256 + tid;
        float w, fw;
        float nw = compute_nw(b, idx, ow, fwt, wmask, smod, style, dvec, &w, &fw);
        s += nw; q += nw*nw;
    }
    sA[tid]=s; sB[tid]=q; __syncthreads();
    for (int st=128; st>0; st>>=1){ if (tid<st){ sA[tid]+=sA[tid+st]; sB[tid]+=sB[tid+st]; } __syncthreads(); }
    if (!tid){ atomicAdd(&acc[2+2*b], sA[0]); atomicAdd(&acc[3+2*b], sB[0]); }
}

// final weight -> f16, layout [b][tap][o][i]   grid (9216, 8)
__global__ void k_w_final(const float* __restrict__ ow, const float* __restrict__ fwt,
                          const float* __restrict__ wmask, const float* __restrict__ smod,
                          const float* __restrict__ style, const float* __restrict__ dvec,
                          const float* __restrict__ acc, _Float16* __restrict__ w16){
    int b = blockIdx.y;
    int idx = blockIdx.x*256 + threadIdx.x;
    const float N = (float)NW_B;
    float s0 = acc[0], q0 = acc[1];
    float mean_ori = s0/N;
    float std_ori  = sqrtf((q0 - s0*s0/N)/(N-1.f));
    float sb_ = acc[2+2*b], qb = acc[3+2*b];
    float mean_w = sb_/N;
    float std_w  = sqrtf((qb - sb_*sb_/N)/(N-1.f));
    float w, fw;
    float nw = compute_nw(b, idx, ow, fwt, wmask, smod, style, dvec, &w, &fw);
    float norm = (nw - mean_w)/std_w*std_ori + mean_ori;
    float fin = (1.f-fw)*w + fw*norm;
    int o = idx / FANIN; int rem = idx - o*FANIN; int i = rem/9; int tap = rem - i*9;
    w16[((size_t)(b*9+tap)*OUT_ + o)*IN_ + i] = (_Float16)fin;
}

// ---------------- WMMA implicit-GEMM conv, double-buffered async staging ----------
// grid (64 rows, 8 OUT-tiles, 8 batches), 256 threads (8 waves)
// A = weights [64 OUT x 32 K], B = padded channels-last image [66 x 32 K] shared
// across the 3 kx taps via an LDS row offset. 48 k-steps = 3 ky x 16 k-blocks.
__global__ __launch_bounds__(256) void k_conv(const _Float16* __restrict__ Xp,
                                              const _Float16* __restrict__ W16,
                                              const float* __restrict__ biasT,
                                              float* __restrict__ out){
    __shared__ __align__(16) _Float16 As[2*ASZ];   // 2 x 3 kx tiles [o_local][k], pad 8
    __shared__ __align__(16) _Float16 Bs[2*BSZ];   // 2 x [x_pad][k], pad 8
    const int y = blockIdx.x, oT = blockIdx.y, b = blockIdx.z;
    const int tid = threadIdx.x;
    const int lane = tid & 31, wv = tid >> 5;
    const int wm = wv & 3;          // M subtile (4 x 16 = 64 OUT)
    const int nhalf = wv >> 2;      // 0/1 -> x base 0/32
    const int row = lane & 15;
    const int hi  = lane >> 4;
    const int aK = hi ? 16 : 0;     // A frag: contiguous K halves
    const int bK = hi ? 16 : 0;     // B frag: contiguous K halves
    const int arow = tid >> 2, aseg = tid & 3;   // staging: 16B per thread
    v8f c0 = {}; v8f c1 = {};

    const _Float16* xbase = Xp + ((size_t)(b*HP + y))*WP*512;
    const _Float16* wbase = W16 + ((size_t)(b*9)*OUT_ + oT*64)*IN_;

    auto stage = [&](int s, int buf){
        const int ky = s >> 4, kb = (s & 15) << 5;
        const _Float16* xrow = xbase + (size_t)ky*WP*512;
        const _Float16* wk   = wbase + (size_t)(ky*3)*OUT_*IN_;
        _Float16* Ab = &As[buf*ASZ];
        _Float16* Bb = &Bs[buf*BSZ];
#pragma unroll
        for (int kx = 0; kx < 3; ++kx){
            async_cp16(&wk[((size_t)kx*OUT_ + arow)*IN_ + kb + aseg*8],
                       &Ab[kx*2560 + arow*40 + aseg*8]);
        }
        async_cp16(&xrow[(size_t)arow*512 + kb + aseg*8], &Bb[arow*40 + aseg*8]);
        if (tid < 8){
            const int c = 64 + (tid >> 2), sg = tid & 3;
            async_cp16(&xrow[(size_t)c*512 + kb + sg*8], &Bb[c*40 + sg*8]);
        }
    };

    stage(0, 0);
    for (int s = 0; s < 48; ++s){
        const int buf = s & 1;
        async_wait0();          // this wave's async writes (tile s) complete
        __syncthreads();        // all waves' writes visible; tile s-1 fully consumed
        if (s + 1 < 48) stage(s + 1, buf ^ 1);   // overlap with compute below
        const int aOfs = buf*ASZ, bOfs = buf*BSZ;
#pragma unroll
        for (int kx = 0; kx < 3; ++kx){
            union { v16h v; uint4 q[2]; } a, b0, b1;
            const int abase = aOfs + kx*2560 + (wm*16 + row)*40 + aK;
            a.q[0] = *(const uint4*)&As[abase];
            a.q[1] = *(const uint4*)&As[abase + 8];
            const int b0base = bOfs + (nhalf*32 + row + kx)*40 + bK;
            b0.q[0] = *(const uint4*)&Bs[b0base];
            b0.q[1] = *(const uint4*)&Bs[b0base + 8];
            const int b1base = bOfs + (nhalf*32 + 16 + row + kx)*40 + bK;
            b1.q[0] = *(const uint4*)&Bs[b1base];
            b1.q[1] = *(const uint4*)&Bs[b1base + 8];
            c0 = __builtin_amdgcn_wmma_f32_16x16x32_f16(false, a.v, false, b0.v,
                                                        (short)0, c0, false, false);
            c1 = __builtin_amdgcn_wmma_f32_16x16x32_f16(false, a.v, false, b1.v,
                                                        (short)0, c1, false, false);
        }
    }
    // store: C/D layout — vgpr v: lanes 0-15 M=v, lanes 16-31 M=v+8; N=lane&15
    const int nlane = lane & 15;
    const int mofs = hi ? 8 : 0;
#pragma unroll
    for (int v=0; v<8; ++v){
        int o = oT*64 + wm*16 + v + mofs;
        float bv = biasT[b*OUT_ + o];
        size_t base = (((size_t)b*OUT_ + o)*H_ + y)*W_;
        out[base + nhalf*32 + nlane]      = c0[v] + bv;
        out[base + nhalf*32 + 16 + nlane] = c1[v] + bv;
    }
}

// ---------------- launcher ----------------

extern "C" void kernel_launch(void* const* d_in, const int* in_sizes, int n_in,
                              void* d_out, int out_size, void* d_ws, size_t ws_size,
                              hipStream_t stream){
    (void)in_sizes; (void)n_in; (void)out_size; (void)ws_size;
    const float* input      = (const float*)d_in[0];
    const float* latent     = (const float*)d_in[1];
    const float* fp         = (const float*)d_in[2];
    const float* ori_weight = (const float*)d_in[3];
    const float* f_weight   = (const float*)d_in[4];
    const float* w_mask     = (const float*)d_in[5];
    const float* b_mask     = (const float*)d_in[6];
    const float* ori_bias   = (const float*)d_in[7];
    const float* W_mlp      = (const float*)d_in[8];
    const float* b_mlp      = (const float*)d_in[9];
    const float* W_fin      = (const float*)d_in[10];
    const float* W_style    = (const float*)d_in[11];
    const float* b_style    = (const float*)d_in[12];
    const float* Wb1        = (const float*)d_in[13];
    const float* Wb2        = (const float*)d_in[14];
    float* out = (float*)d_out;

    char* ws = (char*)d_ws;
    size_t off = 0;
    auto take = [&](size_t bytes)->char*{
        char* p = ws + off;
        off = (off + bytes + 255) & ~((size_t)255);
        return p;
    };
    const size_t xp_elems = (size_t)B_*HP*WP*512;          // 17,842,176 halfs
    _Float16* Xp   = (_Float16*)take(xp_elems*sizeof(_Float16));
    _Float16* W16  = (_Float16*)take((size_t)B_*9*OUT_*IN_*sizeof(_Float16));
    float* smod    = (float*)take(B_*IN_*sizeof(float));
    float* h1      = (float*)take(B_*128*sizeof(float));
    float* t1      = (float*)take(B_*OUT_*sizeof(float));
    float* style   = (float*)take((size_t)B_*9216*sizeof(float));
    float* biasT   = (float*)take(B_*OUT_*sizeof(float));
    float* dvec    = (float*)take(B_*OUT_*sizeof(float));
    float* acc     = (float*)take(32*sizeof(float));

    k_init<<<1, 32, 0, stream>>>(acc);
    k_zero<<<(unsigned)(xp_elems/(8*256)), 256, 0, stream>>>((uint4*)Xp); // 8712 blocks
    k_pad_cvt<<<dim3(64, 8), 256, 0, stream>>>(input, Xp);
    k_small1<<<36, 256, 0, stream>>>(latent, W_mlp, b_mlp, fp, W_fin, Wb1, smod, h1, t1);
    k_small2<<<304, 256, 0, stream>>>(h1, W_style, b_style, t1, Wb2, ori_bias, b_mask,
                                      style, biasT);
    k_base_stats<<<1024, 256, 0, stream>>>(ori_weight, acc);
    k_demod<<<4096, 256, 0, stream>>>(ori_weight, smod, dvec);
    k_w_stats<<<dim3(288, 8), 256, 0, stream>>>(ori_weight, f_weight, w_mask, smod,
                                                style, dvec, acc);
    k_w_final<<<dim3(9216, 8), 256, 0, stream>>>(ori_weight, f_weight, w_mask, smod,
                                                 style, dvec, acc, W16);
    k_conv<<<dim3(64, 8, 8), 256, 0, stream>>>(Xp, W16, biasT, out);
}
